// GCNSiamese2_8624294331071
// MI455X (gfx1250) — matrix operand order repaired and checked by reading, not compiled, not run on hardware
//
#include <hip/hip_runtime.h>
#include <hip/hip_bf16.h>

typedef __attribute__((ext_vector_type(16))) _Float16 v16h;
typedef __attribute__((ext_vector_type(8)))  float    v8f;

#define HDIM 128          // hidden width
#define KDIM 256          // GEMM K = 128 (agg, padded) + 128 (root, padded)
#define WPACK 32768       // packed weight halves per layer: 8 kt * 8 nt * 32 lanes * 16

static inline int cdiv(int a, int b) { return (a + b - 1) / b; }

// ---------------------------------------------------------------- utilities
__global__ void gcs_zero_kernel(float* __restrict__ p, int n) {
    int i = blockIdx.x * blockDim.x + threadIdx.x;
    if (i < n) p[i] = 0.f;
}

// pad raw x [N,F] fp32 -> h [N,128] (cols >= F zeroed)
__global__ void gcs_padx_kernel(const float* __restrict__ x, float* __restrict__ h,
                                int nrows, int F) {
    int idx = blockIdx.x * blockDim.x + threadIdx.x;
    if (idx >= nrows * HDIM) return;
    int r = idx >> 7, c = idx & (HDIM - 1);
    h[idx] = (c < F) ? x[r * F + c] : 0.f;
}

// Pack [W_rel(pad 128) ; W_root(pad 128)] fp32 -> f16 in the exact per-lane
// B-fragment layout of V_WMMA_F32_16X16X32_F16:
//   lane 0-15 : column n = nt*16+lane, K-local = j<8 ? j : 8+j      (j = 0..15)
//   lane 16-31: column n = nt*16+lane-16, K-local = j<8 ? 8+j : 16+j
// Wp[((kt*8+nt)*32+lane)*16 + j]
__global__ void gcs_packw_kernel(const float* __restrict__ Wr, const float* __restrict__ Ws,
                                 _Float16* __restrict__ Wp, int Kin) {
    int t = blockIdx.x * blockDim.x + threadIdx.x;
    if (t >= WPACK) return;
    int j    = t & 15;
    int lane = (t >> 4) & 31;
    int nt   = (t >> 9) & 7;
    int kt   = t >> 12;
    int half = lane >> 4;
    int n    = nt * 16 + (lane & 15);
    int k    = kt * 32 + (half << 3) + ((j & 8) << 1) + (j & 7);   // logical K in [0,256)
    float v;
    if (k < 128) v = (k < Kin) ? Wr[k * HDIM + n] : 0.f;
    else { int k2 = k - 128; v = (k2 < Kin) ? Ws[k2 * HDIM + n] : 0.f; }
    Wp[t] = (_Float16)v;
}

// ------------------------------------------------- edge scatter (wave/edge)
__global__ __launch_bounds__(256)
void gcs_scatter_kernel(const float* __restrict__ hsrc, const int* __restrict__ ei,
                        float* __restrict__ agg, int nE) {
    int wave = (blockIdx.x * blockDim.x + threadIdx.x) >> 5;
    int lane = threadIdx.x & 31;
    if (wave >= nE) return;
    int s = ei[wave];            // edge_index[0][e]
    int d = ei[nE + wave];       // edge_index[1][e]
    const float* xs = hsrc + (size_t)s * HDIM;
    float*       ad = agg  + (size_t)d * HDIM;
#pragma unroll
    for (int i = 0; i < 4; ++i)
        atomicAdd(&ad[lane + i * 32], xs[lane + i * 32]);
}

// ------------------------------------------------------------- WMMA GEMM
// y[N,128] = [agg | h] @ Wp + bias ; A built on the fly (fp32->f16), W in LDS.
// A phase is fully hoisted: all 32 global_load_b128 (8 k-steps x 4 chunks per
// lane) are issued up front and converted into 8 resident v16h fragments, so
// the WMMA loop is a pure LDS -> XDL stream with no global loads, no loadcnt
// waits, and no WAR hazards on the A operand registers.
__global__ __launch_bounds__(256)
void gcs_gemm_wmma_kernel(const float* __restrict__ agg, const float* __restrict__ hin,
                          const _Float16* __restrict__ Wp, const float* __restrict__ bias,
                          float* __restrict__ y, int nrows) {
    __shared__ __align__(32) _Float16 lW[WPACK];   // 64 KB packed weights
    const int tid = threadIdx.x;
    {   // cooperative stage: 4096 float4
        const float4* src = (const float4*)Wp;
        float4*       dst = (float4*)lW;
        for (int i = tid; i < WPACK * 2 / 16; i += 256) dst[i] = src[i];
    }
    __syncthreads();

    const int wave  = tid >> 5;
    const int lane  = tid & 31;
    const int half  = lane >> 4;          // A: K-half select ; C: row-half select
    const int m     = lane & 15;
    const int rbase = blockIdx.x * 128 + wave * 16;
    const int r     = rbase + m;
    const int rc    = (r < nrows) ? r : (nrows - 1);   // clamped row for loads

    const float* arow = agg + (size_t)rc * HDIM;
    const float* hrow = hin + (size_t)rc * HDIM;

    // ---- A phase: load the lane's full 128-float A footprint, cvt to f16.
    // Fragment kt: lane half 0 holds K {0..7,16..23}, half 1 holds {8..15,24..31}.
    v16h af[8];
#pragma unroll
    for (int kt = 0; kt < KDIM / 32; ++kt) {
        const float* srow = (kt < 4) ? arow : hrow;    // uniform per (unrolled) kt
        const int kb = (kt & 3) * 32 + (half << 3);
        float4 c0 = *(const float4*)(srow + kb);
        float4 c1 = *(const float4*)(srow + kb + 4);
        float4 c2 = *(const float4*)(srow + kb + 16);
        float4 c3 = *(const float4*)(srow + kb + 20);
        af[kt][0]  = (_Float16)c0.x;  af[kt][1]  = (_Float16)c0.y;
        af[kt][2]  = (_Float16)c0.z;  af[kt][3]  = (_Float16)c0.w;
        af[kt][4]  = (_Float16)c1.x;  af[kt][5]  = (_Float16)c1.y;
        af[kt][6]  = (_Float16)c1.z;  af[kt][7]  = (_Float16)c1.w;
        af[kt][8]  = (_Float16)c2.x;  af[kt][9]  = (_Float16)c2.y;
        af[kt][10] = (_Float16)c2.z;  af[kt][11] = (_Float16)c2.w;
        af[kt][12] = (_Float16)c3.x;  af[kt][13] = (_Float16)c3.y;
        af[kt][14] = (_Float16)c3.z;  af[kt][15] = (_Float16)c3.w;
    }

    v8f acc[8];
#pragma unroll
    for (int nt = 0; nt < 8; ++nt)
#pragma unroll
        for (int v = 0; v < 8; ++v) acc[nt][v] = 0.f;

    // ---- main loop: pure LDS reads + WMMA.
#pragma unroll
    for (int kt = 0; kt < KDIM / 32; ++kt) {
        v16h bfr[8];
#pragma unroll
        for (int nt = 0; nt < 8; ++nt)
            bfr[nt] = *((const v16h*)(lW + ((size_t)(kt * 8 + nt) * 32 + lane) * 16));
#pragma unroll
        for (int nt = 0; nt < 8; ++nt)
            acc[nt] = __builtin_amdgcn_wmma_f32_16x16x32_f16(
                false, af[kt], false, bfr[nt], (short)0, acc[nt], false, false);
    }

    // C/D layout: col n = nt*16 + (lane&15); VGPR v -> row v + 8*half
#pragma unroll
    for (int nt = 0; nt < 8; ++nt) {
        int n = nt * 16 + m;
        float bv = bias[n];
#pragma unroll
        for (int v = 0; v < 8; ++v) {
            int rr = rbase + (half ? v + 8 : v);
            if (rr < nrows) y[(size_t)rr * HDIM + n] = acc[nt][v] + bv;
        }
    }
}

// ------------------------------------------------------------ BN over nodes
__global__ __launch_bounds__(128)
void gcs_stats_kernel(const float* __restrict__ y, float* __restrict__ sum,
                      float* __restrict__ sumsq, int nrows) {
    int c = threadIdx.x;
    int r0 = blockIdx.x * 256;
    int rend = min(r0 + 256, nrows);
    float s = 0.f, q = 0.f;
    for (int r = r0; r < rend; ++r) {
        float v = y[(size_t)r * HDIM + c];
        s += v; q += v * v;
    }
    atomicAdd(&sum[c], s);
    atomicAdd(&sumsq[c], q);
}

__global__ void gcs_bnrelu_kernel(float* __restrict__ y, const float* __restrict__ sum,
                                  const float* __restrict__ sumsq,
                                  const float* __restrict__ gamma,
                                  const float* __restrict__ beta, int nrows) {
    int idx = blockIdx.x * blockDim.x + threadIdx.x;
    if (idx >= nrows * HDIM) return;
    int c = idx & (HDIM - 1);
    float inv = 1.f / (float)nrows;
    float mu  = sum[c] * inv;
    float var = sumsq[c] * inv - mu * mu;
    float sc  = gamma[c] * rsqrtf(var + 1e-5f);
    float sh  = beta[c] - mu * sc;
    float v = y[idx] * sc + sh;
    y[idx] = v > 0.f ? v : 0.f;
}

// ------------------------------------------------------- gate MLP 128->32->1
__global__ __launch_bounds__(256)
void gcs_gate_kernel(const float* __restrict__ h, const float* __restrict__ Wg1,
                     const float* __restrict__ bg1, const float* __restrict__ Wg2,
                     const float* __restrict__ bg2, float* __restrict__ gate, int nrows) {
    int lane = threadIdx.x & 31;
    int node = (blockIdx.x * blockDim.x + threadIdx.x) >> 5;
    if (node >= nrows) return;
    const float* hr = h + (size_t)node * HDIM;
    float h0 = hr[lane], h1 = hr[lane + 32], h2 = hr[lane + 64], h3 = hr[lane + 96];
    float acc = bg1[lane];                      // lane = hidden unit j (0..31)
    for (int k = 0; k < 32; ++k) {
        float a0 = __shfl(h0, k, 32);
        float a1 = __shfl(h1, k, 32);
        float a2 = __shfl(h2, k, 32);
        float a3 = __shfl(h3, k, 32);
        acc += a0 * Wg1[(k)      * 32 + lane];
        acc += a1 * Wg1[(k + 32) * 32 + lane];
        acc += a2 * Wg1[(k + 64) * 32 + lane];
        acc += a3 * Wg1[(k + 96) * 32 + lane];
    }
    float t = acc > 0.f ? acc : 0.f;
    float p = t * Wg2[lane];
#pragma unroll
    for (int off = 16; off; off >>= 1) p += __shfl_xor(p, off, 32);
    if (lane == 0) {
        float g = p + bg2[0];
        gate[node] = g > 0.f ? g : 0.f;
    }
}

// ------------------------------------------------- attention softmax pooling
__global__ void gcs_gmax_kernel(const float* __restrict__ gate, const int* __restrict__ batch,
                                float* __restrict__ gmax, int nrows) {
    int i = blockIdx.x * blockDim.x + threadIdx.x;
    if (i >= nrows) return;
    // gate >= 0 (ReLU) so int-bit compare == float compare; gmax pre-zeroed
    atomicMax((int*)&gmax[batch[i]], __float_as_int(gate[i]));
}

__global__ void gcs_eg_kernel(const float* __restrict__ gate, const int* __restrict__ batch,
                              const float* __restrict__ gmax, float* __restrict__ eg,
                              float* __restrict__ denom, int nrows) {
    int i = blockIdx.x * blockDim.x + threadIdx.x;
    if (i >= nrows) return;
    int b = batch[i];
    float e = __expf(gate[i] - gmax[b]);
    eg[i] = e;
    atomicAdd(&denom[b], e);
}

__global__ __launch_bounds__(256)
void gcs_pool_kernel(const float* __restrict__ h, const float* __restrict__ eg,
                     const float* __restrict__ denom, const int* __restrict__ batch,
                     float* __restrict__ epool, int nrows) {
    int lane = threadIdx.x & 31;
    int node = (blockIdx.x * blockDim.x + threadIdx.x) >> 5;
    if (node >= nrows) return;
    int b = batch[node];
    float w = eg[node] / denom[b];
    const float* hr = h + (size_t)node * HDIM;
#pragma unroll
    for (int i = 0; i < 4; ++i) {
        int c = lane + i * 32;
        atomicAdd(&epool[b * HDIM + c], w * hr[c]);
    }
}

// ------------------------------------------------- final MLP on |e1-e2|
__global__ __launch_bounds__(64)
void gcs_final_kernel(const float* __restrict__ e1, const float* __restrict__ e2,
                      const float* __restrict__ Wf1, const float* __restrict__ bf1,
                      const float* __restrict__ Wf2, const float* __restrict__ bf2,
                      float* __restrict__ out) {
    __shared__ float hid[64];
    int b = blockIdx.x, j = threadIdx.x;
    float acc = bf1[j];
    for (int k = 0; k < HDIM; ++k) {
        float d = fabsf(e1[b * HDIM + k] - e2[b * HDIM + k]);
        acc += d * Wf1[k * 64 + j];
    }
    hid[j] = acc > 0.f ? acc : 0.f;
    __syncthreads();
    if (j == 0) {
        float s = bf2[0];
        for (int q = 0; q < 64; ++q) s += hid[q] * Wf2[q];
        out[b] = s;
    }
}

// =======================================================================
extern "C" void kernel_launch(void* const* d_in, const int* in_sizes, int n_in,
                              void* d_out, int out_size, void* d_ws, size_t ws_size,
                              hipStream_t stream) {
    (void)n_in; (void)out_size; (void)ws_size;
    const int N = in_sizes[4];              // batch1 length
    const int F = in_sizes[0] / N;          // 38
    const int E = in_sizes[2] / 2;          // edges per side
    const int B = 32;

    const float* x[2]    = {(const float*)d_in[0], (const float*)d_in[1]};
    const int*   ei[2]   = {(const int*)d_in[2], (const int*)d_in[3]};
    const int*   bat[2]  = {(const int*)d_in[4], (const int*)d_in[5]};
    const float* Wr[3]   = {(const float*)d_in[6],  (const float*)d_in[11], (const float*)d_in[16]};
    const float* Ws[3]   = {(const float*)d_in[7],  (const float*)d_in[12], (const float*)d_in[17]};
    const float* bb[3]   = {(const float*)d_in[8],  (const float*)d_in[13], (const float*)d_in[18]};
    const float* gg[3]   = {(const float*)d_in[9],  (const float*)d_in[14], (const float*)d_in[19]};
    const float* be[3]   = {(const float*)d_in[10], (const float*)d_in[15], (const float*)d_in[20]};
    const float* Wg1 = (const float*)d_in[21]; const float* bg1 = (const float*)d_in[22];
    const float* Wg2 = (const float*)d_in[23]; const float* bg2 = (const float*)d_in[24];
    const float* Wf1 = (const float*)d_in[25]; const float* bf1 = (const float*)d_in[26];
    const float* Wf2 = (const float*)d_in[27]; const float* bf2 = (const float*)d_in[28];

    // ---- workspace layout (256B aligned slabs)
    char* wsb = (char*)d_ws;
    size_t off = 0;
    auto alloc = [&](size_t bytes) { char* p = wsb + off; off += (bytes + 255) & ~(size_t)255; return p; };
    float*     bufA   = (float*)alloc((size_t)N * HDIM * 4);
    float*     bufB   = (float*)alloc((size_t)N * HDIM * 4);
    float*     agg    = (float*)alloc((size_t)N * HDIM * 4);
    _Float16*  WpAll  = (_Float16*)alloc((size_t)3 * WPACK * 2);
    float*     colsum = (float*)alloc(HDIM * 4);
    float*     colsq  = (float*)alloc(HDIM * 4);
    float*     gate   = (float*)alloc((size_t)N * 4);
    float*     eg     = (float*)alloc((size_t)N * 4);
    float*     gmax   = (float*)alloc(B * 4);
    float*     denom  = (float*)alloc(B * 4);
    float*     epool0 = (float*)alloc(B * HDIM * 4);
    float*     epool1 = (float*)alloc(B * HDIM * 4);

    const int Kin[3] = {F, HDIM, HDIM};
    for (int l = 0; l < 3; ++l)
        gcs_packw_kernel<<<cdiv(WPACK, 256), 256, 0, stream>>>(Wr[l], Ws[l],
                                                               WpAll + (size_t)l * WPACK, Kin[l]);

    const int nelem   = N * HDIM;
    const int gElem   = cdiv(nelem, 256);
    const int gGemm   = cdiv(N, 128);
    const int gWaveN  = cdiv(N, 8);        // wave-per-node kernels, 8 waves/block
    const int gWaveE  = cdiv(E, 8);        // wave-per-edge

    for (int s = 0; s < 2; ++s) {
        float* hA = bufA; float* hB = bufB;
        gcs_padx_kernel<<<gElem, 256, 0, stream>>>(x[s], hA, N, F);
        for (int l = 0; l < 3; ++l) {
            gcs_zero_kernel<<<gElem, 256, 0, stream>>>(agg, nelem);
            gcs_scatter_kernel<<<gWaveE, 256, 0, stream>>>(hA, ei[s], agg, E);
            gcs_zero_kernel<<<1, 256, 0, stream>>>(colsum, HDIM);
            gcs_zero_kernel<<<1, 256, 0, stream>>>(colsq, HDIM);
            gcs_gemm_wmma_kernel<<<gGemm, 256, 0, stream>>>(agg, hA,
                    WpAll + (size_t)l * WPACK, bb[l], hB, N);
            gcs_stats_kernel<<<cdiv(N, 256), 128, 0, stream>>>(hB, colsum, colsq, N);
            gcs_bnrelu_kernel<<<gElem, 256, 0, stream>>>(hB, colsum, colsq, gg[l], be[l], N);
            float* t = hA; hA = hB; hB = t;
        }
        // hA now holds h3 (post conv_out BN+ReLU)
        float* epool = (s == 0) ? epool0 : epool1;
        gcs_gate_kernel<<<gWaveN, 256, 0, stream>>>(hA, Wg1, bg1, Wg2, bg2, gate, N);
        gcs_zero_kernel<<<1, 256, 0, stream>>>(gmax, B);
        gcs_zero_kernel<<<1, 256, 0, stream>>>(denom, B);
        gcs_zero_kernel<<<cdiv(B * HDIM, 256), 256, 0, stream>>>(epool, B * HDIM);
        gcs_gmax_kernel<<<cdiv(N, 256), 256, 0, stream>>>(gate, bat[s], gmax, N);
        gcs_eg_kernel<<<cdiv(N, 256), 256, 0, stream>>>(gate, bat[s], gmax, eg, denom, N);
        gcs_pool_kernel<<<gWaveN, 256, 0, stream>>>(hA, eg, denom, bat[s], epool, N);
    }

    gcs_final_kernel<<<B, 64, 0, stream>>>(epool0, epool1, Wf1, bf1, Wf2, bf2, (float*)d_out);
}
